// BatchedExpertMoEDispatch_17935783428806
// MI455X (gfx1250) — matrix operand order repaired and checked by reading, not compiled, not run on hardware
//
#include <hip/hip_runtime.h>
#include <hip/hip_bf16.h>

// ---------------------------------------------------------------------------
// MoE FFN (top-2, E=8) for gfx1250: grouped GEMM per expert, bf16 WMMA with
// f32 accumulation, fused gate/up/silu/down using 320KB WGP LDS.
// ---------------------------------------------------------------------------

#define TT 4096
#define HH 1024
#define FF 2048
#define EE 8
#define KK 2

#define TM 32                   // tokens per tile
#define XSTRIDE (HH + 8)        // bf16 elems, pad -> conflict-free ds_load_b128
#define ASTRIDE (FF + 8)
#define SMEM_BYTES ((TM * XSTRIDE + TM * ASTRIDE) * 2 + TM * 4 + TM * 4)

typedef __attribute__((ext_vector_type(16))) __bf16 bf16x16;
typedef __attribute__((ext_vector_type(8)))  float  v8f;
typedef __attribute__((ext_vector_type(4)))  unsigned int   u32x4;
typedef __attribute__((ext_vector_type(4)))  unsigned short u16x4;
typedef __attribute__((ext_vector_type(4)))  float  f32x4;

union Frag16 { bf16x16 v; u32x4 u[2]; };

__device__ __forceinline__ unsigned short f32_to_bf16(float f) {
  unsigned int u = __builtin_bit_cast(unsigned int, f);
  unsigned int r = u + 0x7FFFu + ((u >> 16) & 1u);   // round-to-nearest-even
  return (unsigned short)(r >> 16);
}

__device__ __forceinline__ v8f wmma_bf16(bf16x16 a, bf16x16 b, v8f c) {
  return __builtin_amdgcn_wmma_f32_16x16x32_bf16(
      /*neg_a=*/false, a, /*neg_b=*/false, b,
      /*c_mod=*/(short)0, c, /*reuse_a=*/false, /*reuse_b=*/false);
}

// ---------------------------------------------------------------------------
// Routing: dedup the two slots per token, build per-expert segments.
// ---------------------------------------------------------------------------
__global__ void route_count(const int* __restrict__ ids, int* __restrict__ counts) {
  int t = blockIdx.x * blockDim.x + threadIdx.x;
  if (t >= TT) return;
  int e0 = ids[2 * t], e1 = ids[2 * t + 1];
  atomicAdd(&counts[e0], 1);
  if (e1 != e0) atomicAdd(&counts[e1], 1);
}

__global__ void route_scan(const int* __restrict__ counts,
                           int* __restrict__ offsets, int* __restrict__ cursors) {
  if (threadIdx.x == 0 && blockIdx.x == 0) {
    int s = 0;
    for (int e = 0; e < EE; ++e) { offsets[e] = s; cursors[e] = s; s += counts[e]; }
    offsets[EE] = s;
  }
}

__global__ void route_fill(const int* __restrict__ ids, const float* __restrict__ wts,
                           int* __restrict__ cursors,
                           int* __restrict__ tok_list, float* __restrict__ coeff_list) {
  int t = blockIdx.x * blockDim.x + threadIdx.x;
  if (t >= TT) return;
  int e0 = ids[2 * t], e1 = ids[2 * t + 1];
  float w0 = wts[2 * t], w1 = wts[2 * t + 1];
  if (e0 == e1) {
    int p = atomicAdd(&cursors[e0], 1);
    tok_list[p] = t; coeff_list[p] = w0 + w1;
  } else {
    int p = atomicAdd(&cursors[e0], 1);
    tok_list[p] = t; coeff_list[p] = w0;
    int q = atomicAdd(&cursors[e1], 1);
    tok_list[q] = t; coeff_list[q] = w1;
  }
}

// ---------------------------------------------------------------------------
// Weight prep: f32 [R][C] -> bf16 [C][R] (N-major so WMMA B-fragments are two
// contiguous 16B loads per lane). Tiled transpose through LDS.
// ---------------------------------------------------------------------------
__global__ void transpose_cvt_bf16(const float* __restrict__ src,
                                   unsigned short* __restrict__ dst, int R, int C) {
  __shared__ unsigned short tile[32][33];
  const size_t eoff = (size_t)blockIdx.z * (size_t)R * (size_t)C;
  int c = blockIdx.x * 32 + threadIdx.x;
  int r0 = blockIdx.y * 32;
  #pragma unroll
  for (int j = 0; j < 32; j += 8) {
    int r = r0 + threadIdx.y + j;
    tile[threadIdx.y + j][threadIdx.x] = f32_to_bf16(src[eoff + (size_t)r * C + c]);
  }
  __syncthreads();
  int rOut = blockIdx.y * 32 + threadIdx.x;   // output column = source row
  int cOut0 = blockIdx.x * 32;
  #pragma unroll
  for (int j = 0; j < 32; j += 8) {
    int cc = cOut0 + threadIdx.y + j;
    dst[eoff + (size_t)cc * R + rOut] = tile[threadIdx.x][threadIdx.y + j];
  }
}

// ---------------------------------------------------------------------------
// Main fused kernel: one block = 32-token tile of one expert segment.
// 8 waves; wave w owns 16-wide N column groups. Phase A: G/U WMMA over K=H,
// silu*u -> Act in LDS (bf16). Phase B: Act @ Wd^T over K=F, scaled atomicAdd.
// ---------------------------------------------------------------------------
__global__ __launch_bounds__(256) void moe_ffn_kernel(
    const float* __restrict__ x,
    const int* __restrict__ tok_list, const float* __restrict__ coeff_list,
    const int* __restrict__ counts, const int* __restrict__ offsets,
    const unsigned short* __restrict__ gateT,   // [E][F][H] bf16
    const unsigned short* __restrict__ upT,     // [E][F][H] bf16
    const unsigned short* __restrict__ downT,   // [E][H][F] bf16
    float* __restrict__ out) {

  const int e    = blockIdx.y;
  const int segS = offsets[e];
  const int segL = counts[e];
  const int base = blockIdx.x * TM;
  if (base >= segL) return;
  const int rows = (segL - base < TM) ? (segL - base) : TM;

  extern __shared__ unsigned char smem_raw[];
  unsigned short* Xs   = (unsigned short*)smem_raw;              // [TM][XSTRIDE]
  unsigned short* Act  = Xs + TM * XSTRIDE;                      // [TM][ASTRIDE]
  int*            tokS = (int*)(Act + TM * ASTRIDE);
  float*          cofS = (float*)(tokS + TM);

  const int tid = threadIdx.x;
  if (tid < TM) {
    if (tid < rows) {
      tokS[tid] = tok_list[segS + base + tid];
      cofS[tid] = coeff_list[segS + base + tid];
    } else {
      tokS[tid] = 0;
      cofS[tid] = 0.0f;
    }
  }
  __syncthreads();

  // -------- load X tile, convert f32 -> bf16 into LDS (zero padded rows) ----
  {
    const int rpt = tid >> 3;          // row 0..31
    const int cpt = tid & 7;           // 8 threads per row
    const int trow = tokS[rpt];
    const float* xr = x + (size_t)trow * HH;
    #pragma unroll 4
    for (int i = 0; i < 32; ++i) {
      int col = (cpt + i * 8) * 4;
      u16x4 h;
      if (rpt < rows) {
        f32x4 v = *(const f32x4*)(xr + col);
        h = (u16x4){f32_to_bf16(v.x), f32_to_bf16(v.y),
                    f32_to_bf16(v.z), f32_to_bf16(v.w)};
      } else {
        h = (u16x4){0, 0, 0, 0};
      }
      *(u16x4*)(Xs + rpt * XSTRIDE + col) = h;
    }
  }
  __syncthreads();

  const int lane   = tid & 31;
  const int wv     = tid >> 5;        // wave 0..7
  const int laneLo = lane & 15;
  const int laneHi = lane >> 4;       // 0/1

  const unsigned short* gE = gateT + (size_t)e * FF * HH;
  const unsigned short* uE = upT   + (size_t)e * FF * HH;
  const unsigned short* dE = downT + (size_t)e * HH * FF;

  // ----------------------- Phase A: gate & up, silu ------------------------
  for (int g = 0; g < 16; ++g) {
    const int n0 = (g * 8 + wv) * 16;               // F-column group
    v8f accG0 = {0,0,0,0,0,0,0,0}, accG1 = {0,0,0,0,0,0,0,0};
    v8f accU0 = {0,0,0,0,0,0,0,0}, accU1 = {0,0,0,0,0,0,0,0};
    const unsigned short* bg = gE + (size_t)(n0 + laneLo) * HH + laneHi * 8;
    const unsigned short* bu = uE + (size_t)(n0 + laneLo) * HH + laneHi * 8;

    #pragma unroll 4
    for (int k0 = 0; k0 < HH; k0 += 32) {
      __builtin_prefetch(bg + k0 + 256, 0, 1);
      __builtin_prefetch(bu + k0 + 256, 0, 1);
      const int ka = k0 + laneHi * 8;
      Frag16 a0, a1, fbg, fbu;
      a0.u[0] = *(const u32x4*)(Xs + (laneLo     ) * XSTRIDE + ka);
      a0.u[1] = *(const u32x4*)(Xs + (laneLo     ) * XSTRIDE + ka + 16);
      a1.u[0] = *(const u32x4*)(Xs + (laneLo + 16) * XSTRIDE + ka);
      a1.u[1] = *(const u32x4*)(Xs + (laneLo + 16) * XSTRIDE + ka + 16);
      fbg.u[0] = *(const u32x4*)(bg + k0);
      fbg.u[1] = *(const u32x4*)(bg + k0 + 16);
      fbu.u[0] = *(const u32x4*)(bu + k0);
      fbu.u[1] = *(const u32x4*)(bu + k0 + 16);
      accG0 = wmma_bf16(a0.v, fbg.v, accG0);
      accG1 = wmma_bf16(a1.v, fbg.v, accG1);
      accU0 = wmma_bf16(a0.v, fbu.v, accU0);
      accU1 = wmma_bf16(a1.v, fbu.v, accU1);
    }

    // silu(g)*u -> Act (bf16). C/D layout: elem j -> row j + laneHi*8, col laneLo.
    #pragma unroll
    for (int j = 0; j < 8; ++j) {
      const int r0 = j + laneHi * 8;
      const int col = n0 + laneLo;
      float g0 = accG0[j], u0 = accU0[j];
      float a0v = (g0 / (1.0f + __expf(-g0))) * u0;
      Act[(size_t)r0 * ASTRIDE + col] = f32_to_bf16(a0v);
      float g1 = accG1[j], u1 = accU1[j];
      float a1v = (g1 / (1.0f + __expf(-g1))) * u1;
      Act[(size_t)(r0 + 16) * ASTRIDE + col] = f32_to_bf16(a1v);
    }
  }
  __syncthreads();

  // ----------------------- Phase B: down projection ------------------------
  for (int g = 0; g < 8; ++g) {
    const int n0 = (g * 8 + wv) * 16;               // H-column group
    v8f acc0 = {0,0,0,0,0,0,0,0}, acc1 = {0,0,0,0,0,0,0,0};
    const unsigned short* bd = dE + (size_t)(n0 + laneLo) * FF + laneHi * 8;

    #pragma unroll 4
    for (int k0 = 0; k0 < FF; k0 += 32) {
      __builtin_prefetch(bd + k0 + 256, 0, 1);
      const int ka = k0 + laneHi * 8;
      Frag16 a0, a1, fbd;
      a0.u[0] = *(const u32x4*)(Act + (laneLo     ) * ASTRIDE + ka);
      a0.u[1] = *(const u32x4*)(Act + (laneLo     ) * ASTRIDE + ka + 16);
      a1.u[0] = *(const u32x4*)(Act + (laneLo + 16) * ASTRIDE + ka);
      a1.u[1] = *(const u32x4*)(Act + (laneLo + 16) * ASTRIDE + ka + 16);
      fbd.u[0] = *(const u32x4*)(bd + k0);
      fbd.u[1] = *(const u32x4*)(bd + k0 + 16);
      acc0 = wmma_bf16(a0.v, fbd.v, acc0);
      acc1 = wmma_bf16(a1.v, fbd.v, acc1);
    }

    #pragma unroll
    for (int j = 0; j < 8; ++j) {
      const int r0 = j + laneHi * 8;
      const int r1 = r0 + 16;
      const int col = n0 + laneLo;
      if (r0 < rows)
        atomicAdd(&out[(size_t)tokS[r0] * HH + col], acc0[j] * cofS[r0]);
      if (r1 < rows)
        atomicAdd(&out[(size_t)tokS[r1] * HH + col], acc1[j] * cofS[r1]);
    }
  }
}

// ---------------------------------------------------------------------------
extern "C" void kernel_launch(void* const* d_in, const int* in_sizes, int n_in,
                              void* d_out, int out_size, void* d_ws, size_t ws_size,
                              hipStream_t stream) {
  const float* x       = (const float*)d_in[0];
  const int*   ids     = (const int*)d_in[1];
  const float* wts     = (const float*)d_in[2];
  const float* gate_w  = (const float*)d_in[3];
  const float* up_w    = (const float*)d_in[4];
  const float* down_w  = (const float*)d_in[5];
  float* out = (float*)d_out;

  // workspace layout
  char* ws = (char*)d_ws;
  int*   counts     = (int*)(ws + 0);
  int*   offsets    = (int*)(ws + 64);
  int*   cursors    = (int*)(ws + 128);
  int*   tok_list   = (int*)(ws + 256);
  float* coeff_list = (float*)(ws + 256 + (size_t)TT * KK * 4);
  size_t wo = 256 + 2 * (size_t)TT * KK * 4;
  wo = (wo + 255) & ~(size_t)255;
  const size_t WBYTES = (size_t)EE * HH * FF * 2;    // bf16 per tensor
  unsigned short* gateT = (unsigned short*)(ws + wo); wo += WBYTES;
  unsigned short* upT   = (unsigned short*)(ws + wo); wo += WBYTES;
  unsigned short* downT = (unsigned short*)(ws + wo); wo += WBYTES;

  hipMemsetAsync(d_ws, 0, 256, stream);                            // counts/cursors
  hipMemsetAsync(d_out, 0, (size_t)TT * HH * sizeof(float), stream);

  // bf16 + transpose weight copies (one-time ~300MB of HBM traffic, then L2-resident)
  transpose_cvt_bf16<<<dim3(FF / 32, HH / 32, EE), dim3(32, 8), 0, stream>>>(
      gate_w, gateT, HH, FF);
  transpose_cvt_bf16<<<dim3(FF / 32, HH / 32, EE), dim3(32, 8), 0, stream>>>(
      up_w, upT, HH, FF);
  transpose_cvt_bf16<<<dim3(HH / 32, FF / 32, EE), dim3(32, 8), 0, stream>>>(
      down_w, downT, FF, HH);

  // routing
  route_count<<<TT / 256, 256, 0, stream>>>(ids, counts);
  route_scan<<<1, 32, 0, stream>>>(counts, offsets, cursors);
  route_fill<<<TT / 256, 256, 0, stream>>>(ids, wts, cursors, tok_list, coeff_list);

  // main fused grouped-GEMM (needs ~198KB dynamic LDS; CDNA5 WGP has 320KB)
  hipFuncSetAttribute((const void*)moe_ffn_kernel,
                      hipFuncAttributeMaxDynamicSharedMemorySize, SMEM_BYTES);
  moe_ffn_kernel<<<dim3(TT / TM, EE), 256, SMEM_BYTES, stream>>>(
      x, tok_list, coeff_list, counts, offsets, gateT, upT, downT, out);
}
